// EstimatorQNNGen275_65481071407711
// MI455X (gfx1250) — matrix operand order repaired and checked by reading, not compiled, not run on hardware
//
#include <hip/hip_runtime.h>
#include <hip/hip_bf16.h>
#include <math.h>

typedef __bf16 bf16_t;
typedef __attribute__((ext_vector_type(8)))  __bf16 v8bf;
typedef __attribute__((ext_vector_type(16))) __bf16 v16bf;
typedef __attribute__((ext_vector_type(8)))  float  v8f;

#define MDIM 512
#define NDIM 4096
#define KDIM 4096
#define HW   128
#define CONVN (127 * 127)

__device__ __forceinline__ bf16_t f2bf(float f) {
    union { float f; unsigned u; } x; x.f = f;
    unsigned r = x.u + 0x7FFFu + ((x.u >> 16) & 1u);   // round-to-nearest-even
    union { unsigned short s; bf16_t b; } y; y.s = (unsigned short)(r >> 16);
    return y.b;
}
__device__ __forceinline__ float bf2f(bf16_t b) {
    union { unsigned short s; bf16_t b; } y; y.b = b;
    union { unsigned u; float f; } x; x.u = ((unsigned)y.s) << 16;
    return x.f;
}

// ---------------- Stage 1: conv 2x2 + sigmoid + mean over 127x127 ----------------
__global__ __launch_bounds__(256)
void conv_mean_kernel(const float* __restrict__ x, const float* __restrict__ cw,
                      const float* __restrict__ cb, float* __restrict__ convout) {
    const int b = blockIdx.x;
    const float w0 = cw[0], w1 = cw[1], w2 = cw[2], w3 = cw[3], bias = cb[0];
    const float* xb = x + (size_t)b * HW * HW;
    float s = 0.f;
    for (int p = threadIdx.x; p < CONVN; p += 256) {
        int i = p / 127, j = p % 127;
        const float* r0 = xb + i * HW + j;
        float z = r0[0] * w0 + r0[1] * w1 + r0[HW] * w2 + r0[HW + 1] * w3 + bias;
        s += 1.f / (1.f + expf(-z));
    }
    __shared__ float red[256];
    red[threadIdx.x] = s; __syncthreads();
    for (int o = 128; o > 0; o >>= 1) {
        if (threadIdx.x < o) red[threadIdx.x] += red[threadIdx.x + o];
        __syncthreads();
    }
    if (threadIdx.x == 0) convout[b] = red[0] * (1.f / (float)CONVN);
}

// ---------------- Stage 2: 1-D attention (EMBED_DIM=1) ----------------
__global__ void attn_kernel(const float* __restrict__ convout, const float* __restrict__ rot,
                            const float* __restrict__ ent, float* __restrict__ attn) {
    __shared__ float c[MDIM];
    const int t = threadIdx.x;
    c[t] = convout[t];
    __syncthreads();
    const float scale = rot[0] * ent[0];
    const float ci = c[t];
    float mx = -3.4e38f;
    for (int j = 0; j < MDIM; ++j) mx = fmaxf(mx, scale * ci * c[j]);
    float se = 0.f, ws = 0.f;
    for (int j = 0; j < MDIM; ++j) {
        float e = expf(scale * ci * c[j] - mx);
        se += e; ws += e * c[j];
    }
    attn[t] = ws / se;
}

// ---------------- Stage 3: layer 1 is rank-1 (fi == 1), elementwise ----------------
__global__ __launch_bounds__(256)
void layer1_kernel(const float* __restrict__ attn, const float* __restrict__ W1,
                   const float* __restrict__ b1, bf16_t* __restrict__ a1b) {
    int idx = blockIdx.x * 256 + threadIdx.x;     // < 512*4096
    int i = idx >> 12, j = idx & 4095;
    float v = tanhf(attn[i] * W1[j] + b1[j]);
    a1b[idx] = f2bf(v);
}

// ---------------- f32 -> bf16 weight conversion ----------------
__global__ __launch_bounds__(256)
void cvt_bf16_kernel(const float* __restrict__ w, bf16_t* __restrict__ wb) {
    int idx = (blockIdx.x * 256 + threadIdx.x) * 4;   // n is a multiple of 4
    float4 v = *(const float4*)(w + idx);
    wb[idx + 0] = f2bf(v.x);
    wb[idx + 1] = f2bf(v.y);
    wb[idx + 2] = f2bf(v.z);
    wb[idx + 3] = f2bf(v.w);
}

// ---------------- Stage 4: bf16 WMMA GEMM: act = tanh(A @ W^T + b) ----------------
// Per wave: one 32x64 C tile (2 M-frags x 4 N-accumulators). Each B fragment is
// reused by 2 WMMAs, each A fragment by 4 -> 8 WMMA : 12 b128 loads per 32-K step.
__global__ __launch_bounds__(256)
void gemm_tanh_bf16(const bf16_t* __restrict__ A, const bf16_t* __restrict__ W,
                    const float* __restrict__ bias, bf16_t* __restrict__ actb,
                    float* __restrict__ actf) {
    const int lane = threadIdx.x & 31;
    const int wave = blockIdx.x * 8 + (threadIdx.x >> 5);   // 1024 waves total
    const int ntiles = NDIM / 64;                            // 64
    const int tm = (wave / ntiles) * 32;
    const int tn = (wave % ntiles) * 64;

    const int mrow0  = tm + (lane & 15);
    const int kbaseA = (lane & 16) ? 8 : 0;     // A frag: K kb..kb+7 and kb+16..kb+23
    const int kbaseB = (lane & 16) ? 8 : 0;     // B frag: K kb..kb+7 and kb+16..kb+23
    const int ncol0  = tn + (lane & 15);

    v8f acc[2][4];
#pragma unroll
    for (int mt = 0; mt < 2; ++mt)
#pragma unroll
        for (int j = 0; j < 4; ++j)
            acc[mt][j] = (v8f){0.f, 0.f, 0.f, 0.f, 0.f, 0.f, 0.f, 0.f};

    const bf16_t* Arow0 = A + (size_t)mrow0 * KDIM;
    const bf16_t* Arow1 = Arow0 + (size_t)16 * KDIM;
    for (int k = 0; k < KDIM; k += 32) {
        v16bf af[2];
#pragma unroll
        for (int mt = 0; mt < 2; ++mt) {
            const bf16_t* Ap = (mt == 0 ? Arow0 : Arow1) + k + kbaseA;
            const v8bf a0 = *(const v8bf*)(Ap);
            const v8bf a1 = *(const v8bf*)(Ap + 16);
#pragma unroll
            for (int t = 0; t < 8; ++t) { af[mt][t] = a0[t]; af[mt][8 + t] = a1[t]; }
        }
#pragma unroll
        for (int j = 0; j < 4; ++j) {
            // B[k][n] = W[n][k] : two 8-wide K chunks of row n of W
            const bf16_t* Wp = W + (size_t)(ncol0 + 16 * j) * KDIM + k + kbaseB;
            const v8bf b0 = *(const v8bf*)(Wp);
            const v8bf b1 = *(const v8bf*)(Wp + 16);
            v16bf bfr;
#pragma unroll
            for (int t = 0; t < 8; ++t) { bfr[t] = b0[t]; bfr[8 + t] = b1[t]; }
            acc[0][j] = __builtin_amdgcn_wmma_f32_16x16x32_bf16(
                false, af[0], false, bfr, (short)0, acc[0][j], false, false);
            acc[1][j] = __builtin_amdgcn_wmma_f32_16x16x32_bf16(
                false, af[1], false, bfr, (short)0, acc[1][j], false, false);
        }
    }

    // epilogue: C layout -> m = r + (lane>=16)*8, n = lane&15 within tile
#pragma unroll
    for (int mt = 0; mt < 2; ++mt) {
        const int mbase = tm + mt * 16 + ((lane >> 4) << 3);
#pragma unroll
        for (int j = 0; j < 4; ++j) {
            const int n = ncol0 + 16 * j;
            const float b = bias[n];
#pragma unroll
            for (int r = 0; r < 8; ++r) {
                float v = tanhf(acc[mt][j][r] + b);
                size_t off = (size_t)(mbase + r) * NDIM + n;
                actb[off] = f2bf(v);
                if (actf) actf[off] = v;
            }
        }
    }
}

// ---------------- Stage 5: pairwise Gram of the 4 activations ----------------
__global__ void zero_gram_kernel(float* __restrict__ gram) {
    if (threadIdx.x < 16) gram[threadIdx.x] = 0.f;
}

__global__ __launch_bounds__(256)
void gram_kernel(const bf16_t* __restrict__ a1, const bf16_t* __restrict__ a2,
                 const bf16_t* __restrict__ a3, const bf16_t* __restrict__ a4,
                 float* __restrict__ gram) {
    const int n = MDIM * NDIM;
    float s[10];
#pragma unroll
    for (int p = 0; p < 10; ++p) s[p] = 0.f;
    for (int idx = blockIdx.x * 256 + threadIdx.x; idx < n; idx += gridDim.x * 256) {
        float v0 = bf2f(a1[idx]), v1 = bf2f(a2[idx]);
        float v2 = bf2f(a3[idx]), v3 = bf2f(a4[idx]);
        s[0] += v0 * v0; s[1] += v1 * v1; s[2] += v2 * v2; s[3] += v3 * v3;
        s[4] += v0 * v1; s[5] += v0 * v2; s[6] += v0 * v3;
        s[7] += v1 * v2; s[8] += v1 * v3; s[9] += v2 * v3;
    }
    __shared__ float red[256];
    for (int p = 0; p < 10; ++p) {
        red[threadIdx.x] = s[p]; __syncthreads();
        for (int o = 128; o > 0; o >>= 1) {
            if (threadIdx.x < o) red[threadIdx.x] += red[threadIdx.x + o];
            __syncthreads();
        }
        if (threadIdx.x == 0) atomicAdd(&gram[p], red[0]);
        __syncthreads();
    }
}

__global__ void finalize_kernel(const float* __restrict__ gram, float* __restrict__ wout) {
    int t = threadIdx.x;
    if (t >= 16) return;
    int l = t >> 2, m = t & 3;
    int lo = l < m ? l : m, hi = l < m ? m : l;
    float d;
    if (l == m) d = gram[l];
    else {
        const int base[3] = {4, 7, 9};
        d = gram[base[lo] + (hi - lo - 1)];
    }
    float nl = sqrtf(gram[l]) + 1e-12f;
    float nm = sqrtf(gram[m]) + 1e-12f;
    float c = d / (nl * nm);
    float fid = c * c;
    wout[l * 4 + m] = (fid >= 0.8f && l != m) ? 1.0f : 0.0f;
}

extern "C" void kernel_launch(void* const* d_in, const int* in_sizes, int n_in,
                              void* d_out, int out_size, void* d_ws, size_t ws_size,
                              hipStream_t stream) {
    const float* x    = (const float*)d_in[0];
    const float* cw   = (const float*)d_in[1];
    const float* cb   = (const float*)d_in[2];
    const float* rot  = (const float*)d_in[3];
    const float* ent  = (const float*)d_in[4];
    const float* W1   = (const float*)d_in[5];
    const float* b1   = (const float*)d_in[6];
    const float* W2   = (const float*)d_in[7];
    const float* b2   = (const float*)d_in[8];
    const float* W3   = (const float*)d_in[9];
    const float* b3   = (const float*)d_in[10];
    const float* W4   = (const float*)d_in[11];
    const float* b4   = (const float*)d_in[12];
    float* outf = (float*)d_out;                        // (512,4096) then 4x4 weights
    float* wgts = outf + (size_t)MDIM * NDIM;

    // workspace carve-up
    char* ws = (char*)d_ws;
    size_t off = 0;
    float* convout = (float*)(ws + off); off += 512 * sizeof(float);
    float* attn    = (float*)(ws + off); off += 512 * sizeof(float);
    float* gram    = (float*)(ws + off); off += 16 * sizeof(float);
    off = (off + 255) & ~(size_t)255;
    const size_t actN = (size_t)MDIM * NDIM;            // 2,097,152
    const size_t wN   = (size_t)NDIM * KDIM;            // 16,777,216
    bf16_t* a1b = (bf16_t*)(ws + off); off += actN * 2;
    bf16_t* a2b = (bf16_t*)(ws + off); off += actN * 2;
    bf16_t* a3b = (bf16_t*)(ws + off); off += actN * 2;
    bf16_t* a4b = (bf16_t*)(ws + off); off += actN * 2;
    bf16_t* w2b = (bf16_t*)(ws + off); off += wN * 2;
    bf16_t* w3b = (bf16_t*)(ws + off); off += wN * 2;
    bf16_t* w4b = (bf16_t*)(ws + off); off += wN * 2;

    // weight conversions (independent of pipeline front)
    const int cvtBlocks = (int)(wN / 4 / 256);          // 16384
    cvt_bf16_kernel<<<cvtBlocks, 256, 0, stream>>>(W2, w2b);
    cvt_bf16_kernel<<<cvtBlocks, 256, 0, stream>>>(W3, w3b);
    cvt_bf16_kernel<<<cvtBlocks, 256, 0, stream>>>(W4, w4b);

    conv_mean_kernel<<<MDIM, 256, 0, stream>>>(x, cw, cb, convout);
    attn_kernel<<<1, MDIM, 0, stream>>>(convout, rot, ent, attn);
    layer1_kernel<<<(int)(actN / 256), 256, 0, stream>>>(attn, W1, b1, a1b);

    // GEMM layers: 1024 waves = 128 blocks x 8 waves, exact cover (EXEC all-1s)
    gemm_tanh_bf16<<<128, 256, 0, stream>>>(a1b, w2b, b2, a2b, nullptr);
    gemm_tanh_bf16<<<128, 256, 0, stream>>>(a2b, w3b, b3, a3b, nullptr);
    gemm_tanh_bf16<<<128, 256, 0, stream>>>(a3b, w4b, b4, a4b, outf);

    zero_gram_kernel<<<1, 32, 0, stream>>>(gram);
    gram_kernel<<<2048, 256, 0, stream>>>(a1b, a2b, a3b, a4b, gram);
    finalize_kernel<<<1, 32, 0, stream>>>(gram, wgts);
}
